// KOSLMSSM_47528108098128
// MI455X (gfx1250) — matrix operand order repaired and checked by reading, not compiled, not run on hardware
//
#include <hip/hip_runtime.h>

// ---------------------------------------------------------------------------
// CDNA5 / gfx1250 fused kernel for the KOSLMSSM cell forward.
// bf16 hi/lo split-precision WMMA (f32_16x16x32_bf16, 3 products) for both
// GEMMs, fully fused per 32-row tile; hidden stays in LDS.
// Wave mapping: each wave owns a unique N-range and BOTH 16-row M-blocks, so
// every B (weight) fragment load feeds two WMMAs -> half the L2 weight traffic.
// ---------------------------------------------------------------------------

typedef __bf16 v16bf __attribute__((ext_vector_type(16)));
typedef float  v8f   __attribute__((ext_vector_type(8)));

#define HDIM   512
#define BATCH  64
#define HID    1536            // 3*H
#define ROWS   32              // rows of (b,i) space per workgroup
#define NC     256             // hidden-dim chunk width
#define NCHUNK (HID / NC)      // 6
#define LDI    (HDIM + 8)      // padded innov row stride (bank-conflict free)
#define LDH    (NC + 8)        // padded hidden row stride

// ---- helpers ---------------------------------------------------------------

__device__ __forceinline__ void split_bf16(float x, __bf16& hi, __bf16& lo) {
    hi = (__bf16)x;                 // RNE
    lo = (__bf16)(x - (float)hi);   // residual
}

__device__ __forceinline__ float sigmoid_f(float x) {
    return 1.0f / (1.0f + __expf(-x));
}

// A fragment (16x32 bf16) from LDS, row-major with stride `stride`.
// Contiguous k order used here (same as gB): lane L holds row M = L%16; K values c0..c0+15 contiguous,
// c0 = (L<16 ? 0 : 16), relative to kbase.
__device__ __forceinline__ v16bf ldsA(const __bf16* row0, int stride, int kbase, int lane) {
    const __bf16* p = row0 + (lane & 15) * stride + kbase + ((lane & 16) ? 16 : 0);
    v16bf a;
    __builtin_memcpy(&a, p, 16);                  // K c0..c0+7      (ds_load_b128)
    __builtin_memcpy((char*)&a + 16, p + 8, 16);  // K c0+8..c0+15   (ds_load_b128)
    return a;
}

// B fragment (32x16 bf16) from global weight stored row-major [n][k] (ld=ldw).
// Contiguous k order used here (same as ldsA): lane L holds column N = L%16; K = kbase + (L<16?0:16) + 0..15
// contiguous -> a single 32-byte per-lane read.
__device__ __forceinline__ v16bf gB(const __bf16* w, int ldw, int nbase, int kbase, int lane) {
    const __bf16* p = w + (size_t)(nbase + (lane & 15)) * ldw + kbase + ((lane & 16) ? 16 : 0);
    v16bf b;
    __builtin_memcpy(&b, p, 32);                  // 2x global_load_b128
    return b;
}

__device__ __forceinline__ v8f wmma_bf16(v16bf a, v16bf b, v8f c) {
    return __builtin_amdgcn_wmma_f32_16x16x32_bf16(
        /*neg_a=*/false, a, /*neg_b=*/false, b,
        /*c_mod=*/(short)0, c, /*reuse_a=*/false, /*reuse_b=*/false);
}

// ---- stage 0: convert W1 [1536x512] and W2 [512x1536] f32 -> bf16 hi/lo ----

__global__ __launch_bounds__(256) void convert_weights_kernel(
    const float* __restrict__ W1, const float* __restrict__ W2,
    __bf16* __restrict__ w1hi, __bf16* __restrict__ w1lo,
    __bf16* __restrict__ w2hi, __bf16* __restrict__ w2lo) {
    const int N1 = HID * HDIM;  // 786432 (same count for W2)
    int idx = blockIdx.x * 256 + threadIdx.x;
    if (idx < N1) {
        __bf16 hi, lo;
        split_bf16(W1[idx], hi, lo);
        w1hi[idx] = hi; w1lo[idx] = lo;
    } else {
        int k = idx - N1;
        __bf16 hi, lo;
        split_bf16(W2[k], hi, lo);
        w2hi[k] = hi; w2lo[k] = lo;
    }
}

// ---- main fused kernel -----------------------------------------------------

__global__ __launch_bounds__(256) void koslmssm_fused_kernel(
    const float* __restrict__ gates,   // [B, 2H]
    const float* __restrict__ c_prev,  // [B, H]
    const float* __restrict__ A,       // [B, H, H]
    const float* __restrict__ At_prev, // [B, H, H]
    const float* __restrict__ b1,      // [3H]
    const float* __restrict__ b2,      // [H]
    const __bf16* __restrict__ w1hi, const __bf16* __restrict__ w1lo,
    const __bf16* __restrict__ w2hi, const __bf16* __restrict__ w2lo,
    float* __restrict__ out) {

    __shared__ __bf16 innovHi[ROWS][LDI];
    __shared__ __bf16 innovLo[ROWS][LDI];
    __shared__ __bf16 hidHi[ROWS][LDH];
    __shared__ __bf16 hidLo[ROWS][LDH];
    __shared__ float zL[ROWS], ML[ROWS], sL[ROWS], cpL[ROWS], rowsum[ROWS];

    const int tid  = threadIdx.x;
    const int lane = tid & 31;
    const int w    = tid >> 5;          // wave id 0..7
    const int rowBase = blockIdx.x * ROWS;

    float* outH = out;                            // h_new  [32768]
    float* outC = out + 32768;                    // c_new  [B*H*H]
    float* outA = out + 32768 + (size_t)BATCH * HDIM * HDIM;  // A_t

    // phase 0: per-row scalars
    if (tid < ROWS) {
        int r = rowBase + tid;
        int b = r >> 9;            // / 512
        int i = r & (HDIM - 1);
        float z = tanhf(gates[b * (2 * HDIM) + i]);
        float M = sigmoid_f(gates[b * (2 * HDIM) + HDIM + i]);
        float cp = c_prev[r];
        zL[tid] = z; ML[tid] = M; cpL[tid] = cp; sL[tid] = cp * M;
        rowsum[tid] = 0.0f;
    }
    __syncthreads();

    // phase 1: innov = z - At_prev * (c_prev*M), split into bf16 hi/lo in LDS
    {
        const float4* At4 = (const float4*)(At_prev + (size_t)rowBase * HDIM);
#pragma unroll
        for (int it = 0; it < 16; ++it) {
            int f   = tid + it * 256;      // 0..4095 float4s over [32 x 512]
            int row = f >> 7;              // /128
            int c4  = (f & 127) * 4;
            float4 v = At4[f];
            float z = zL[row], s = sL[row];
            float e0 = z - v.x * s, e1 = z - v.y * s;
            float e2 = z - v.z * s, e3 = z - v.w * s;
            __bf16 h0, l0, h1, l1, h2, l2, h3, l3;
            split_bf16(e0, h0, l0); split_bf16(e1, h1, l1);
            split_bf16(e2, h2, l2); split_bf16(e3, h3, l3);
            innovHi[row][c4 + 0] = h0; innovHi[row][c4 + 1] = h1;
            innovHi[row][c4 + 2] = h2; innovHi[row][c4 + 3] = h3;
            innovLo[row][c4 + 0] = l0; innovLo[row][c4 + 1] = l1;
            innovLo[row][c4 + 2] = l2; innovLo[row][c4 + 3] = l3;
        }
    }
    __syncthreads();

    // Persistent K_t accumulators: this wave owns columns (w*4+t)*16..+15 for
    // t=0..3, and BOTH M-blocks (rows 0..15 and 16..31).
    // acc2[t*2 + mbi] = tile (rows mbi*16..+15, cols (w*4+t)*16..+15).
    v8f acc2[8];
#pragma unroll
    for (int t = 0; t < 8; ++t) acc2[t] = (v8f){0, 0, 0, 0, 0, 0, 0, 0};

    for (int c = 0; c < NCHUNK; ++c) {
        // ---- GEMM1: hidden chunk [32 x NC] = innov[32x512] @ W1^T cols ----
        // Wave owns N-tiles nt = w*2, w*2+1; each B fragment feeds both
        // M-blocks (two WMMAs per product term).
#pragma unroll
        for (int tt = 0; tt < 2; ++tt) {
            int nt   = w * 2 + tt;         // 0..15 within chunk (unique per wave)
            int ncol = c * NC + nt * 16;   // global hidden column base
            v8f acc0 = (v8f){0, 0, 0, 0, 0, 0, 0, 0};   // rows 0..15
            v8f acc1 = (v8f){0, 0, 0, 0, 0, 0, 0, 0};   // rows 16..31
#pragma unroll
            for (int kb = 0; kb < HDIM / 32; ++kb) {
                int kbase = kb * 32;
                v16bf bhi = gB(w1hi, HDIM, ncol, kbase, lane);
                v16bf blo = gB(w1lo, HDIM, ncol, kbase, lane);
                v16bf a0h = ldsA(&innovHi[0][0],  LDI, kbase, lane);
                v16bf a0l = ldsA(&innovLo[0][0],  LDI, kbase, lane);
                v16bf a1h = ldsA(&innovHi[16][0], LDI, kbase, lane);
                v16bf a1l = ldsA(&innovLo[16][0], LDI, kbase, lane);
                acc0 = wmma_bf16(a0h, bhi, acc0);
                acc0 = wmma_bf16(a0h, blo, acc0);
                acc0 = wmma_bf16(a0l, bhi, acc0);
                acc1 = wmma_bf16(a1h, bhi, acc1);
                acc1 = wmma_bf16(a1h, blo, acc1);
                acc1 = wmma_bf16(a1l, bhi, acc1);
            }
            // bias + relu, re-split to bf16 hi/lo into LDS
            int colL  = nt * 16 + (lane & 15);    // chunk-local column
            float bias = b1[c * NC + colL];
            int rhalf = (lane >> 4) * 8;
#pragma unroll
            for (int v = 0; v < 8; ++v) {
                int lr0 = v + rhalf;          // M-block 0
                float x0 = fmaxf(acc0[v] + bias, 0.0f);
                __bf16 hi0, lo0;
                split_bf16(x0, hi0, lo0);
                hidHi[lr0][colL] = hi0;
                hidLo[lr0][colL] = lo0;
                int lr1 = 16 + v + rhalf;     // M-block 1
                float x1 = fmaxf(acc1[v] + bias, 0.0f);
                __bf16 hi1, lo1;
                split_bf16(x1, hi1, lo1);
                hidHi[lr1][colL] = hi1;
                hidLo[lr1][colL] = lo1;
            }
        }
        __syncthreads();

        // ---- GEMM2 partial: K_t += hidden[32xNC] @ W2^T k-slice ----
#pragma unroll
        for (int kb = 0; kb < NC / 32; ++kb) {
            int kbase = kb * 32;
            v16bf a0h = ldsA(&hidHi[0][0],  LDH, kbase, lane);
            v16bf a0l = ldsA(&hidLo[0][0],  LDH, kbase, lane);
            v16bf a1h = ldsA(&hidHi[16][0], LDH, kbase, lane);
            v16bf a1l = ldsA(&hidLo[16][0], LDH, kbase, lane);
            int kg = c * NC + kbase;
#pragma unroll
            for (int t = 0; t < 4; ++t) {
                int n = (w * 4 + t) * 16;     // unique K_t column tile per wave
                v16bf bhi = gB(w2hi, HID, n, kg, lane);
                v16bf blo = gB(w2lo, HID, n, kg, lane);
                acc2[t * 2 + 0] = wmma_bf16(a0h, bhi, acc2[t * 2 + 0]);
                acc2[t * 2 + 0] = wmma_bf16(a0h, blo, acc2[t * 2 + 0]);
                acc2[t * 2 + 0] = wmma_bf16(a0l, bhi, acc2[t * 2 + 0]);
                acc2[t * 2 + 1] = wmma_bf16(a1h, bhi, acc2[t * 2 + 1]);
                acc2[t * 2 + 1] = wmma_bf16(a1h, blo, acc2[t * 2 + 1]);
                acc2[t * 2 + 1] = wmma_bf16(a1l, bhi, acc2[t * 2 + 1]);
            }
        }
        __syncthreads();   // hid LDS reused by next chunk
    }

    // ---- epilogue: A_t, c_new, row-sums for h_new ----
#pragma unroll
    for (int t = 0; t < 4; ++t) {
        int j = (w * 4 + t) * 16 + (lane & 15);   // column in [0,512)
        float bias = b2[j];
        int rhalf = (lane >> 4) * 8;
#pragma unroll
        for (int mbi = 0; mbi < 2; ++mbi) {
#pragma unroll
            for (int v = 0; v < 8; ++v) {
                int lr = mbi * 16 + v + rhalf;
                int r  = rowBase + lr;
                int i  = r & (HDIM - 1);
                float M = ML[lr], z = zL[lr], cp = cpL[lr];
                float kt = acc2[t * 2 + mbi][v] + bias;
                float diag = (i == j) ? 1.0f : 0.0f;
                size_t gidx = (size_t)r * HDIM + j;
                float at = sigmoid_f(diag - kt * M) * A[gidx];
                float cn = at * cp + kt * z;
                outA[gidx] = at;
                outC[gidx] = cn;
                // reduce cn across the 16 lanes sharing this row (N direction)
                float red = cn;
                red += __shfl_xor(red, 8, 16);
                red += __shfl_xor(red, 4, 16);
                red += __shfl_xor(red, 2, 16);
                red += __shfl_xor(red, 1, 16);
                if ((lane & 15) == 0) atomicAdd(&rowsum[lr], red);
            }
        }
    }
    __syncthreads();
    if (tid < ROWS) {
        outH[rowBase + tid] = rowsum[tid] * ML[tid] * (1.0f / (float)HDIM);
    }
}

// ---------------------------------------------------------------------------

extern "C" void kernel_launch(void* const* d_in, const int* in_sizes, int n_in,
                              void* d_out, int out_size, void* d_ws, size_t ws_size,
                              hipStream_t stream) {
    const float* gates   = (const float*)d_in[0];
    const float* c_prev  = (const float*)d_in[1];
    const float* A       = (const float*)d_in[2];
    const float* At_prev = (const float*)d_in[3];
    // d_in[4] = W_k (unused), d_in[5] = b_k (unused)
    const float* W1 = (const float*)d_in[6];
    const float* b1 = (const float*)d_in[7];
    const float* W2 = (const float*)d_in[8];
    const float* b2 = (const float*)d_in[9];
    float* out = (float*)d_out;

    const int NW = HID * HDIM;  // 786432 elements per weight matrix
    __bf16* w1hi = (__bf16*)d_ws;
    __bf16* w1lo = w1hi + NW;
    __bf16* w2hi = w1lo + NW;
    __bf16* w2lo = w2hi + NW;

    // stage 0: f32 -> bf16 hi/lo weight planes (6 MB of d_ws)
    convert_weights_kernel<<<(2 * NW) / 256, 256, 0, stream>>>(
        W1, W2, w1hi, w1lo, w2hi, w2lo);

    // fused forward: 1024 row-tiles of 32 rows, 256 threads (8 wave32) each
    koslmssm_fused_kernel<<<(BATCH * HDIM) / ROWS, 256, 0, stream>>>(
        gates, c_prev, A, At_prev, b1, b2, w1hi, w1lo, w2hi, w2lo, out);
}
